// CosineSimHashDecoder_80788334837893
// MI455X (gfx1250) — matrix operand order repaired and checked by reading, not compiled, not run on hardware
//
#include <hip/hip_runtime.h>
#include <hip/hip_bf16.h>

// ---------------------------------------------------------------------------
// CosineSimHashDecoder for MI455X (gfx1250, wave32, WMMA)
//   Phase 1: LSH band codes (ballot-packed) + f16-normalized rows -> workspace
//   Phase 2: 128x128-tiled zn @ zn^T via v_wmma_f32_16x16x32_f16, masked
//            non-temporal store; LDS staging via global_load_async_to_lds.
// ---------------------------------------------------------------------------

#define N_NODES 8192
#define D_FEAT  128
#define CODE_WORDS 16          // 64 bands * 8 bits = 16 u32 words per node
#define TILE 128
#define KCHUNK 64
#define LDS_STRIDE 72          // 64 + 8 halves pad: conflict-free, 16B aligned

typedef _Float16 v16h __attribute__((ext_vector_type(16)));
typedef _Float16 h8   __attribute__((ext_vector_type(8)));
typedef _Float16 h4   __attribute__((ext_vector_type(4)));
typedef float    v8f  __attribute__((ext_vector_type(8)));

static __device__ __forceinline__ unsigned ballot32(bool pred) {
#if __has_builtin(__builtin_amdgcn_ballot_w32)
  return __builtin_amdgcn_ballot_w32(pred);
#else
  return (unsigned)__ballot(pred);
#endif
}

// --- gfx1250 async global->LDS copies (ASYNCcnt path), with safe fallback ---
#if __has_builtin(__builtin_amdgcn_global_load_async_to_lds_b128) && \
    __has_builtin(__builtin_amdgcn_global_load_async_to_lds_b32)
#define HAVE_ASYNC_COPY 1
// builtin param types (from clang diagnostics):
//   b128: (AS1 int __vector(4)*, AS3 int __vector(4)*, Imm i32, Imm i32)
//   b32 : (AS1 int*,             AS3 int*,             Imm i32, Imm i32)
typedef int vec4i __attribute__((vector_size(16)));
typedef __attribute__((address_space(1))) vec4i* g128p;
typedef __attribute__((address_space(3))) vec4i* l128p;
typedef __attribute__((address_space(1))) int*   g32p;
typedef __attribute__((address_space(3))) int*   l32p;
static __device__ __forceinline__ void copy_b128_to_lds(void* lds, const void* g) {
  __builtin_amdgcn_global_load_async_to_lds_b128(
      (g128p)(unsigned long long)g, (l128p)(unsigned long long)lds, 0, 0);
}
static __device__ __forceinline__ void copy_b32_to_lds(void* lds, const void* g) {
  __builtin_amdgcn_global_load_async_to_lds_b32(
      (g32p)(unsigned long long)g, (l32p)(unsigned long long)lds, 0, 0);
}
static __device__ __forceinline__ void wait_async0() {
#if __has_builtin(__builtin_amdgcn_s_wait_asynccnt)
  __builtin_amdgcn_s_wait_asynccnt(0);
#else
  asm volatile("s_wait_asynccnt 0" ::: "memory");
#endif
}
#else
#define HAVE_ASYNC_COPY 0
static __device__ __forceinline__ void copy_b128_to_lds(void* lds, const void* g) {
  *(uint4*)lds = *(const uint4*)g;
}
static __device__ __forceinline__ void copy_b32_to_lds(void* lds, const void* g) {
  *(unsigned*)lds = *(const unsigned*)g;
}
static __device__ __forceinline__ void wait_async0() {}
#endif

// ---------------------------------------------------------------------------
// Kernel 1: one wave per node. Computes:
//   codes[n][w] for w=0..15 : word w packs band codes 4w..4w+3 (byte each)
//   zn[n][d] = (half)(z[n][d] / ||z[n]||)
// Sign of z@H is invariant to positive scaling, so raw z is used for bits.
// ---------------------------------------------------------------------------
__global__ __launch_bounds__(256)
void lsh_hash_kernel(const float* __restrict__ z, const float* __restrict__ H,
                     _Float16* __restrict__ zn, unsigned* __restrict__ codes) {
  __shared__ float zsh[8][D_FEAT];

  const unsigned tid  = threadIdx.x;
  const unsigned wv   = tid >> 5;
  const unsigned lane = tid & 31;
  const unsigned n    = blockIdx.x * 8 + wv;

  // load this node's row (4 floats per lane) and stage to LDS
  const float4 zv = *(const float4*)&z[(size_t)n * D_FEAT + lane * 4];
  *(float4*)&zsh[wv][lane * 4] = zv;
  __syncthreads();

  // 16 projection accumulators: lane handles planes p = t*32 + lane
  float acc[16];
#pragma unroll
  for (int t = 0; t < 16; ++t) acc[t] = 0.0f;
  float ss = 0.0f;

  for (int d = 0; d < D_FEAT; ++d) {
    const float zd = zsh[wv][d];
    ss += zd * zd;
    const float* Hd = H + (size_t)d * 512 + lane;
#pragma unroll
    for (int t = 0; t < 16; ++t) acc[t] += zd * Hd[t * 32];
  }

  // normalized row in f16
  const float inv = rsqrtf(ss);
  h4 hv;
  hv.x = (_Float16)(zv.x * inv);
  hv.y = (_Float16)(zv.y * inv);
  hv.z = (_Float16)(zv.z * inv);
  hv.w = (_Float16)(zv.w * inv);
  *(h4*)&zn[(size_t)n * D_FEAT + lane * 4] = hv;

  // ballot t gives planes t*32..t*32+31 = bands 4t..4t+3 packed byte-wise:
  // bit index (lane) = (band%4)*8 + bit_within_band  -> word == packed codes
  unsigned myword = 0;
#pragma unroll
  for (int t = 0; t < 16; ++t) {
    const unsigned m = ballot32(acc[t] > 0.0f);
    if (lane == (unsigned)t) myword = m;
  }
  if (lane < CODE_WORDS) codes[(size_t)n * CODE_WORDS + lane] = myword;
}

// any byte of x equal to zero?
static __device__ __forceinline__ bool has_zero_byte(unsigned x) {
  return ((x - 0x01010101u) & ~x & 0x80808080u) != 0u;
}

// ---------------------------------------------------------------------------
// Kernel 2: 128x128 output tile per block (grid 64x64), 8 waves.
// Wave w computes rows R+16w..R+16w+15 against all 128 tile columns:
// 8 accumulator fragments, K=128 as 2 LDS chunks x 2 wmma k-steps.
// ---------------------------------------------------------------------------
__global__ __launch_bounds__(256)
void sim_mask_kernel(const _Float16* __restrict__ zn,
                     const unsigned* __restrict__ codes,
                     float* __restrict__ out) {
  __shared__ _Float16 ldsA[TILE * LDS_STRIDE];
  __shared__ _Float16 ldsB[TILE * LDS_STRIDE];
  __shared__ unsigned ldsCA[TILE * CODE_WORDS];
  __shared__ unsigned ldsCB[TILE * CODE_WORDS];

  const unsigned tid  = threadIdx.x;
  const unsigned wv   = tid >> 5;
  const unsigned lane = tid & 31;
  const unsigned l15  = lane & 15;
  const bool     hi   = lane >= 16;
  const unsigned R = blockIdx.y * TILE;
  const unsigned C = blockIdx.x * TILE;

  // stage band-code panels (2048 words per side, 8 per thread) -- async
#pragma unroll
  for (int i = 0; i < 8; ++i) {
    const unsigned idx = tid + i * 256;                 // row = idx/16, word = idx%16
    copy_b32_to_lds(&ldsCA[idx],
                    &codes[(size_t)(R + (idx >> 4)) * CODE_WORDS + (idx & 15)]);
    copy_b32_to_lds(&ldsCB[idx],
                    &codes[(size_t)(C + (idx >> 4)) * CODE_WORDS + (idx & 15)]);
  }

  v8f acc[8];
#pragma unroll
  for (int t = 0; t < 8; ++t) acc[t] = (v8f){};

  for (int kc = 0; kc < D_FEAT; kc += KCHUNK) {
    __syncthreads();   // previous chunk fully consumed before overwrite
    // stage zn panels: 128 rows x 64 halves per side = 1024 16B chunks, 4/thread
#pragma unroll
    for (int i = 0; i < 4; ++i) {
      const unsigned c   = tid + i * 256;
      const unsigned row = c >> 3;
      const unsigned off = (c & 7) * 8;
      copy_b128_to_lds(&ldsA[row * LDS_STRIDE + off],
                       &zn[(size_t)(R + row) * D_FEAT + kc + off]);
      copy_b128_to_lds(&ldsB[row * LDS_STRIDE + off],
                       &zn[(size_t)(C + row) * D_FEAT + kc + off]);
    }
    wait_async0();     // this wave's async LDS writes complete
    __syncthreads();   // all waves' writes visible

#pragma unroll
    for (int ks = 0; ks < KCHUNK; ks += 32) {
      // A fragment (16x32 f16): lo lanes K=[ks..+15], hi lanes K=[ks+16..+31]
      const unsigned ar = wv * 16 + l15;
      const _Float16* pa = &ldsA[ar * LDS_STRIDE + ks + (hi ? 16 : 0)];
      const h8 alo = *(const h8*)pa;
      const h8 ahi = *(const h8*)(pa + 8);
      const v16h afrag = __builtin_shufflevector(
          alo, ahi, 0, 1, 2, 3, 4, 5, 6, 7, 8, 9, 10, 11, 12, 13, 14, 15);
#pragma unroll
      for (int ct = 0; ct < 8; ++ct) {
        // B fragment (32x16 f16): lane's column, contiguous 16 halves
        const unsigned bc = ct * 16 + l15;
        const _Float16* pb = &ldsB[bc * LDS_STRIDE + ks + (hi ? 16 : 0)];
        const h8 blo = *(const h8*)pb;
        const h8 bhi = *(const h8*)(pb + 8);
        const v16h bfrag = __builtin_shufflevector(
            blo, bhi, 0, 1, 2, 3, 4, 5, 6, 7, 8, 9, 10, 11, 12, 13, 14, 15);
        acc[ct] = __builtin_amdgcn_wmma_f32_16x16x32_f16(
            false, afrag, false, bfrag, (short)0, acc[ct], false, false);
      }
    }
  }

  // epilogue: C/D layout -> lane column = l15, rows = vgpr + (hi?8:0)
#pragma unroll
  for (int ct = 0; ct < 8; ++ct) {
    const unsigned jloc = ct * 16 + l15;
    const unsigned gj   = C + jloc;
    unsigned cc[CODE_WORDS];
#pragma unroll
    for (int w = 0; w < CODE_WORDS; ++w) cc[w] = ldsCB[jloc * CODE_WORDS + w];
#pragma unroll
    for (int v = 0; v < 8; ++v) {
      const unsigned iloc = wv * 16 + v + (hi ? 8 : 0);
      const unsigned gi   = R + iloc;
      const float s = acc[ct][v];
      float val = 0.0f;
      if (gi == gj) {
        val = 1.0f;                       // identity on diagonal
      } else if (s > 0.75f) {             // dist < 0.25  <=>  S > 0.75
        bool coll = false;
#pragma unroll
        for (int w = 0; w < CODE_WORDS; ++w)
          coll |= has_zero_byte(cc[w] ^ ldsCA[iloc * CODE_WORDS + w]);
        if (coll) val = s;
      }
      // write-once 256MB result: keep it out of the cache hierarchy
      __builtin_nontemporal_store(val, &out[(size_t)gi * N_NODES + gj]);
    }
  }
}

// ---------------------------------------------------------------------------
extern "C" void kernel_launch(void* const* d_in, const int* in_sizes, int n_in,
                              void* d_out, int out_size, void* d_ws, size_t ws_size,
                              hipStream_t stream) {
  const float* z = (const float*)d_in[0];      // [8192,128] f32
  const float* H = (const float*)d_in[1];      // [128,512]  f32
  // d_in[2] = edge_index (unused by the reference forward pass)
  float* out = (float*)d_out;                  // [8192,8192] f32

  _Float16* zn = (_Float16*)d_ws;                                  // 2 MB
  unsigned* codes =
      (unsigned*)((char*)d_ws + (size_t)N_NODES * D_FEAT * sizeof(_Float16)); // 512 KB

  lsh_hash_kernel<<<N_NODES / 8, 256, 0, stream>>>(z, H, zn, codes);

  dim3 grid(N_NODES / TILE, N_NODES / TILE);   // 64 x 64
  sim_mask_kernel<<<grid, 256, 0, stream>>>(zn, codes, out);
}
